// PHRLinear_15290083574085
// MI455X (gfx1250) — matrix operand order repaired and checked by reading, not compiled, not run on hardware
//
#include <hip/hip_runtime.h>

typedef __attribute__((ext_vector_type(16))) __bf16 v16bf;
typedef __attribute__((ext_vector_type(8)))  __bf16 v8bf;
typedef __attribute__((ext_vector_type(4)))  __bf16 v4bf;
typedef __attribute__((ext_vector_type(2)))  __bf16 v2bf;
typedef __attribute__((ext_vector_type(8)))  float  v8f;

#define IN_F   4096
#define OUT_F  11008
#define MTOT   256
#define BM     128
#define BN     64
#define BK     64
#define KSTEPS (IN_F / BK)   // 64
#define XPITCH 72            // bf16 elems; 144B row: 16B aligned, 36-bank stride (conflict-free mod 16)
#define WPITCH 72

__global__ __launch_bounds__(256)
void phr_linear_wmma_bf16(const __bf16* __restrict__ x,
                          const int*    __restrict__ Wp,
                          const __bf16* __restrict__ Wf,
                          const float*  __restrict__ lut,
                          const __bf16* __restrict__ bias,
                          float*        __restrict__ out)
{
    __shared__ float  lut_s[256];
    __shared__ __bf16 xs[2][BM * XPITCH];   // double-buffered x tile (M-major)
    __shared__ __bf16 ws[2][BN * WPITCH];   // double-buffered dequant W tile (N-major)

    const int tid  = threadIdx.x;        // 0..255
    const int wave = tid >> 5;           // 0..7  -> 16-row M sub-tile
    const int lane = tid & 31;
    const int half = lane >> 4;
    const int l16  = lane & 15;

    const int m_base = blockIdx.x * BM;  // M fast -> adjacent blocks share weight tiles in L2
    const int n_base = blockIdx.y * BN;

    lut_s[tid] = lut[tid];
    __syncthreads();                     // lut_s visible before first gather

    // dequant assignment: thread -> 4 consecutive n, k row-pair (pass 1 adds +32)
    const int dq_n = (tid & 15) * 4;
    const int dq_k = (tid >> 4) * 2;

    // in-flight WEIGHT tile registers only (24 VGPRs), loaded one K-step ahead.
    // x tile is L2-resident (2 MB total) -> staged synchronously, no held regs.
    int4  wpa[2], wpb[2];
    v4bf  wfa[2], wfb[2];

    auto load_weights = [&](int kt) {
        const int k0 = kt * BK;
        #pragma unroll
        for (int p = 0; p < 2; ++p) {
            const size_t g = (size_t)(k0 + dq_k + p * 32) * OUT_F + n_base + dq_n;
            wpa[p] = *(const int4*)(Wp + g);
            wpb[p] = *(const int4*)(Wp + g + OUT_F);
            wfa[p] = *(const v4bf*)(Wf + g);
            wfb[p] = *(const v4bf*)(Wf + g + OUT_F);
        }
    };

    auto stage = [&](int buf, int kt) {
        const int k0 = kt * BK;
        // x tile: 128 rows x 64 bf16 (128B/row) = 4 x 16B chunks / thread (transient regs)
        #pragma unroll
        for (int p = 0; p < 4; ++p) {
            const int c = tid + p * 256, row = c >> 3, cc = c & 7;
            const uint4 v = *(const uint4*)(x + (size_t)(m_base + row) * IN_F + k0 + cc * 8);
            *(uint4*)(&xs[buf][row * XPITCH + cc * 8]) = v;
        }
        // dequant held weight regs -> N-major LDS tile, packed b32 stores along k.
        // W_p values are guaranteed in [0,256) by construction -> no mask needed.
        #pragma unroll
        for (int p = 0; p < 2; ++p) {
            const int k = dq_k + p * 32;
            #pragma unroll
            for (int j = 0; j < 4; ++j) {
                const float w0 = lut_s[((const int*)&wpa[p])[j]] + (float)wfa[p][j];
                const float w1 = lut_s[((const int*)&wpb[p])[j]] + (float)wfb[p][j];
                v2bf pkv;
                pkv[0] = (__bf16)w0;
                pkv[1] = (__bf16)w1;
                *(v2bf*)(&ws[buf][(dq_n + j) * WPITCH + k]) = pkv;  // k even -> b32 store
            }
        }
    };

    v8f acc0 = {}, acc1 = {}, acc2 = {}, acc3 = {};

    load_weights(0);
    int buf = 0;

    #pragma unroll 1
    for (int kt = 0; kt < KSTEPS; ++kt) {
        stage(buf, kt);         // x tile + dequant of weights loaded last iteration
        __syncthreads();        // single barrier per K-step (double-buffered)

        if (kt + 1 < KSTEPS)
            load_weights(kt + 1);   // next weight tile overlaps this tile's WMMAs

        if (kt + 2 < KSTEPS) {      // distance-2 speculative prefetch (global_prefetch_b8)
            const size_t g = (size_t)((kt + 2) * BK + dq_k) * OUT_F + n_base + dq_n;
            __builtin_prefetch(Wp + g, 0, 0);
            __builtin_prefetch(Wf + g, 0, 0);
        }

        // ---- compute: 2 K-halves x 4 N sub-tiles = 8 WMMAs
        #pragma unroll
        for (int kh = 0; kh < 2; ++kh) {
            const __bf16* ap = &xs[buf][(wave * 16 + l16) * XPITCH + kh * 32 + half * 8];
            const v8bf alo = *(const v8bf*)ap;
            const v8bf ahi = *(const v8bf*)(ap + 16);
            const v16bf a = __builtin_shufflevector(alo, ahi,
                              0,1,2,3,4,5,6,7, 8,9,10,11,12,13,14,15);
            #pragma unroll
            for (int ns = 0; ns < 4; ++ns) {
                const __bf16* bp = &ws[buf][(ns * 16 + l16) * WPITCH + kh * 32 + half * 8];
                const v8bf blo = *(const v8bf*)bp;
                const v8bf bhi = *(const v8bf*)(bp + 16);
                const v16bf b = __builtin_shufflevector(blo, bhi,
                                  0,1,2,3,4,5,6,7, 8,9,10,11,12,13,14,15);
                if      (ns == 0) acc0 = __builtin_amdgcn_wmma_f32_16x16x32_bf16(false, a, false, b, (short)0, acc0, false, false);
                else if (ns == 1) acc1 = __builtin_amdgcn_wmma_f32_16x16x32_bf16(false, a, false, b, (short)0, acc1, false, false);
                else if (ns == 2) acc2 = __builtin_amdgcn_wmma_f32_16x16x32_bf16(false, a, false, b, (short)0, acc2, false, false);
                else              acc3 = __builtin_amdgcn_wmma_f32_16x16x32_bf16(false, a, false, b, (short)0, acc3, false, false);
            }
        }

        buf ^= 1;
    }

    // ---- epilogue: D VGPR r -> row (half*8 + r), col = l16; add bias, store f32
    #pragma unroll
    for (int ns = 0; ns < 4; ++ns) {
        const int gn   = n_base + ns * 16 + l16;
        const float bv = (float)bias[gn];
        float* orow = out + (size_t)(m_base + wave * 16 + half * 8) * OUT_F + gn;
        const v8f accv = (ns == 0) ? acc0 : (ns == 1) ? acc1 : (ns == 2) ? acc2 : acc3;
        #pragma unroll
        for (int r = 0; r < 8; ++r)
            orow[(size_t)r * OUT_F] = accv[r] + bv;
    }
}

extern "C" void kernel_launch(void* const* d_in, const int* in_sizes, int n_in,
                              void* d_out, int out_size, void* d_ws, size_t ws_size,
                              hipStream_t stream) {
    const __bf16* x    = (const __bf16*)d_in[0];
    const int*    Wp   = (const int*)d_in[1];
    const __bf16* Wf   = (const __bf16*)d_in[2];
    const float*  lutp = (const float*)d_in[3];
    const __bf16* bias = (const __bf16*)d_in[4];
    float* out = (float*)d_out;

    dim3 grid(MTOT / BM, OUT_F / BN);   // (2, 172)
    phr_linear_wmma_bf16<<<grid, dim3(256), 0, stream>>>(x, Wp, Wf, lutp, bias, out);
}
